// SpGAT_25305947308670
// MI455X (gfx1250) — compile-verified, matrix operand order checked by reading, not run on hardware
//
#include <hip/hip_runtime.h>
#include <hip/hip_bf16.h>

// ---------------------------------------------------------------------------
// SpGAT for MI455X (gfx1250): edge-gather GEMMs on v_wmma_f32_16x16x32_f16,
// segment sums via hardware f32 global atomics, wave32 layout throughout.
// ---------------------------------------------------------------------------

typedef __attribute__((ext_vector_type(16))) _Float16 v16h;
typedef __attribute__((ext_vector_type(8)))  float    v8f;

#define N_NODES   30000
#define N_EDGES1  150000
#define N_EDGESN  30000
#define N_EDGES   180000          // 150000 + 30000
#define N_STRIPS  11250           // 180000 / 16
#define ALPHA_LR  0.2f

// ---- workspace layout (bytes) ---------------------------------------------
static constexpr size_t X16_OFF   = 0;                       // 30000*128*2
static constexpr size_t X16_SZ    = (size_t)N_NODES * 128 * 2;
static constexpr size_t H16_OFF   = X16_OFF + X16_SZ;        // 30000*224*2
static constexpr size_t H16_SZ    = (size_t)N_NODES * 224 * 2;
static constexpr size_t EE_OFF    = H16_OFF + H16_SZ;        // max(180000*128, 180000*224)*2
static constexpr size_t EE_SZ     = (size_t)N_EDGES * 224 * 2;
static constexpr size_t ACC_OFF   = EE_OFF + EE_SZ;          // 30000*200*4
static constexpr size_t ACC_SZ    = (size_t)N_NODES * 200 * 4;
static constexpr size_t RS1_OFF   = ACC_OFF + ACC_SZ;        // 30000*2*4
static constexpr size_t RS1_SZ    = (size_t)N_NODES * 2 * 4;
static constexpr size_t RS2_OFF   = RS1_OFF + RS1_SZ;        // 30000*4
static constexpr size_t RS2_SZ    = (size_t)N_NODES * 4;
static constexpr size_t B1_OFF    = RS2_OFF + RS2_SZ;        // 224*384*2
static constexpr size_t B1_SZ     = (size_t)224 * 384 * 2;
static constexpr size_t B2_OFF    = B1_OFF + B1_SZ;          // 208*672*2
static constexpr size_t B2_SZ     = (size_t)208 * 672 * 2;
static constexpr size_t REL16_OFF = B2_OFF + B2_SZ;          // 208*128*2
static constexpr size_t REL16_SZ  = (size_t)208 * 128 * 2;
static constexpr size_t WT16_OFF  = REL16_OFF + REL16_SZ;    // 208*128*2
static constexpr size_t WT16_SZ   = (size_t)208 * 128 * 2;
static constexpr size_t A2P1_OFF  = WT16_OFF + WT16_SZ;      // 224*4 (pad 1024)
static constexpr size_t A2P2_OFF  = A2P1_OFF + 1024;         // 208*4 (pad 1024)
static constexpr size_t SRC_OFF   = A2P2_OFF + 1024;         // 180000*4
static constexpr size_t DST_OFF   = SRC_OFF + (size_t)N_EDGES * 4;

// ---- fragment helpers ------------------------------------------------------
union FragU { uint4 u[2]; v16h h; };

// CDNA5 16-bit 16x32 A/B fragment: lane<16 holds K {kb..kb+7, kb+16..kb+23},
// lane>=16 holds K {kb+8..kb+15, kb+24..kb+31}; two contiguous 16B loads.
__device__ __forceinline__ v16h load_frag(const _Float16* __restrict__ row,
                                          int kb, int hi) {
  FragU f;
  f.u[0] = *(const uint4*)(row + kb + hi * 8);
  f.u[1] = *(const uint4*)(row + kb + 16 + hi * 8);
  return f.h;
}

__device__ __forceinline__ void atomAddF(float* p, float v) {
  __hip_atomic_fetch_add(p, v, __ATOMIC_RELAXED, __HIP_MEMORY_SCOPE_AGENT);
}

__device__ __forceinline__ float elu1(float v) {
  return v > 0.f ? v : expm1f(v);
}

// ---------------------------------------------------------------------------
// setup / conversion kernels
// ---------------------------------------------------------------------------
__global__ void sg_build_edges(const int* __restrict__ el,
                               const int* __restrict__ eln,
                               int* __restrict__ srcc, int* __restrict__ dstc) {
  int i = blockIdx.x * 256 + threadIdx.x;
  if (i >= N_EDGES) return;
  if (i < N_EDGES1) { srcc[i] = el[i]; dstc[i] = el[N_EDGES1 + i]; }
  else { int e = i - N_EDGES1; srcc[i] = eln[e]; dstc[i] = eln[N_EDGESN + e]; }
}

__global__ void sg_build_x16(const float* __restrict__ x, _Float16* __restrict__ x16) {
  int i = blockIdx.x * 256 + threadIdx.x;
  if (i >= N_NODES * 128) return;
  int node = i >> 7, k = i & 127;
  float v = (k < 100) ? x[node * 100 + k] : 0.f;
  x16[i] = (_Float16)v;
}

__global__ void sg_build_ee1(const float* __restrict__ eemb,
                             const float* __restrict__ rel,
                             const int* __restrict__ etn,
                             _Float16* __restrict__ ee16) {
  int i = blockIdx.x * 256 + threadIdx.x;
  if (i >= N_EDGES * 128) return;
  int e = i >> 7, k = i & 127;
  float v = 0.f;
  if (k < 100) {
    if (e < N_EDGES1) v = eemb[(size_t)e * 100 + k];
    else {
      int en = e - N_EDGES1;
      int t0 = etn[en * 2], t1 = etn[en * 2 + 1];
      v = rel[t0 * 100 + k] + rel[t1 * 100 + k];
    }
  }
  ee16[i] = (_Float16)v;
}

__global__ void sg_build_B1(const float* __restrict__ a_h, _Float16* __restrict__ B1) {
  int i = blockIdx.x * 256 + threadIdx.x;
  if (i >= 224 * 384) return;
  int n = i / 384, k = i % 384;
  int h = n / 112, nl = n % 112;
  int seg = k / 128, kl = k % 128;
  float v = (nl < 100 && kl < 100) ? a_h[((size_t)(h * 100 + nl)) * 300 + seg * 100 + kl] : 0.f;
  B1[i] = (_Float16)v;
}

__global__ void sg_build_a2p1(const float* __restrict__ a2_h, float* __restrict__ a2p) {
  int i = blockIdx.x * 256 + threadIdx.x;
  if (i >= 224) return;
  int h = i / 112, nl = i % 112;
  a2p[i] = (nl < 100) ? a2_h[h * 100 + nl] : 0.f;
}

__global__ void sg_build_relWT(const float* __restrict__ rel, const float* __restrict__ W,
                               _Float16* __restrict__ rel16, _Float16* __restrict__ WT16) {
  int i = blockIdx.x * 256 + threadIdx.x;
  if (i >= 208 * 128) return;
  int r = i / 128, k = i % 128;
  rel16[i] = (_Float16)((r < 200 && k < 100) ? rel[r * 100 + k] : 0.f);
  WT16[i]  = (_Float16)((r < 200 && k < 100) ? W[k * 200 + r] : 0.f);
}

__global__ void sg_build_B2(const float* __restrict__ a_o, _Float16* __restrict__ B2) {
  int i = blockIdx.x * 256 + threadIdx.x;
  if (i >= 208 * 672) return;
  int n = i / 672, k = i % 672;
  int seg = k / 224, kl = k % 224;
  float v = (n < 200 && kl < 200) ? a_o[(size_t)n * 600 + seg * 200 + kl] : 0.f;
  B2[i] = (_Float16)v;
}

__global__ void sg_build_a2p2(const float* __restrict__ a2_o, float* __restrict__ a2p) {
  int i = blockIdx.x * 256 + threadIdx.x;
  if (i >= 208) return;
  a2p[i] = (i < 200) ? a2_o[i] : 0.f;
}

// out_relation = relation_embed @ W  (200x100 @ 100x200), WMMA tiles 13x13
__global__ __launch_bounds__(256) void sg_relW_gemm(const _Float16* __restrict__ rel16,
                                                    const _Float16* __restrict__ WT16,
                                                    float* __restrict__ outrel) {
  const int lane = threadIdx.x & 31;
  const int tile = blockIdx.x * 8 + (threadIdx.x >> 5);
  if (tile >= 13 * 13) return;
  const int ti = tile / 13, tj = tile % 13;
  const int hl = lane & 15, hi = lane >> 4;
  v8f c = {};
#pragma unroll
  for (int k = 0; k < 4; ++k) {
    v16h a = load_frag(rel16 + (size_t)(ti * 16 + hl) * 128, k * 32, hi);
    v16h b = load_frag(WT16 + (size_t)(tj * 16 + hl) * 128, k * 32, hi);
    c = __builtin_amdgcn_wmma_f32_16x16x32_f16(false, a, false, b, (short)0, c, false, false);
  }
#pragma unroll
  for (int r = 0; r < 8; ++r) {
    int m = ti * 16 + r + hi * 8;
    int n = tj * 16 + hl;
    if (m < 200 && n < 200) outrel[m * 200 + n] = c[r];
  }
}

__global__ void sg_build_ee2(const float* __restrict__ orel,
                             const int* __restrict__ etype,
                             const int* __restrict__ etn,
                             _Float16* __restrict__ ee16) {
  int i = blockIdx.x * 256 + threadIdx.x;
  if (i >= N_EDGES * 224) return;
  int e = i / 224, k = i % 224;
  float v = 0.f;
  if (k < 200) {
    if (e < N_EDGES1) v = orel[(size_t)etype[e] * 200 + k];
    else {
      int en = e - N_EDGES1;
      int t0 = etn[en * 2], t1 = etn[en * 2 + 1];
      v = orel[(size_t)t0 * 200 + k] + orel[(size_t)t1 * 200 + k];
    }
  }
  ee16[i] = (_Float16)v;
}

// ---------------------------------------------------------------------------
// Layer-1 edge kernel: one wave per 16-edge strip, Npad=224 (2 heads x 112),
// K = 3 segments x 128 -> 12 WMMA k-steps, 14 N tiles.
// ---------------------------------------------------------------------------
__global__ __launch_bounds__(256) void sg_l1_edge(
    const _Float16* __restrict__ x16, const _Float16* __restrict__ ee16,
    const _Float16* __restrict__ B1, const float* __restrict__ a2p,
    const int* __restrict__ src, const int* __restrict__ dst,
    float* __restrict__ acc, float* __restrict__ rowsum) {
  const int lane = threadIdx.x & 31;
  const int strip = blockIdx.x * 8 + (threadIdx.x >> 5);
  if (strip >= N_STRIPS) return;                  // wave-uniform: EXEC stays full
  const int e0 = strip * 16;
  const int hl = lane & 15, hi = lane >> 4;
  const int eA = e0 + hl;
  const int sA = src[eA], dA = dst[eA];
  const _Float16* arow[3] = { x16 + (size_t)sA * 128,
                              x16 + (size_t)dA * 128,
                              ee16 + (size_t)eA * 128 };
  v8f accf[14] = {};
#pragma unroll
  for (int seg = 0; seg < 3; ++seg) {
    const _Float16* ar = arow[seg];
#pragma unroll
    for (int kc = 0; kc < 4; ++kc) {
      v16h af = load_frag(ar, kc * 32, hi);
      const int kg = (seg * 4 + kc) * 32;
#pragma unroll
      for (int t = 0; t < 14; ++t) {
        v16h bf = load_frag(B1 + (size_t)(t * 16 + hl) * 384, kg, hi);
        accf[t] = __builtin_amdgcn_wmma_f32_16x16x32_f16(false, af, false, bf,
                                                         (short)0, accf[t], false, false);
      }
    }
  }
  int srcs[8];
#pragma unroll
  for (int r = 0; r < 8; ++r) srcs[r] = src[e0 + r + hi * 8];

#pragma unroll
  for (int h = 0; h < 2; ++h) {
    float p[8];
#pragma unroll
    for (int r = 0; r < 8; ++r) p[r] = 0.f;
#pragma unroll
    for (int tt = 0; tt < 7; ++tt) {
      const int t = h * 7 + tt;
      const float a2v = a2p[t * 16 + hl];
#pragma unroll
      for (int r = 0; r < 8; ++r) p[r] += accf[t][r] * a2v;
    }
#pragma unroll
    for (int m = 1; m < 16; m <<= 1)
#pragma unroll
      for (int r = 0; r < 8; ++r) p[r] += __shfl_xor(p[r], m, 32);
    float ev[8];
#pragma unroll
    for (int r = 0; r < 8; ++r) {
      float z = p[r];
      float lr = z > 0.f ? z : ALPHA_LR * z;     // leaky_relu
      ev[r] = expf(-lr);                          // exp(-leaky)
    }
    if (hl == 0) {
#pragma unroll
      for (int r = 0; r < 8; ++r) atomAddF(&rowsum[srcs[r] * 2 + h], ev[r]);
    }
#pragma unroll
    for (int tt = 0; tt < 7; ++tt) {
      const int t = h * 7 + tt;
      const int cl = tt * 16 + hl;
      if (cl < 100) {
        const int cg = h * 100 + cl;
#pragma unroll
        for (int r = 0; r < 8; ++r)
          atomAddF(&acc[(size_t)srcs[r] * 200 + cg], ev[r] * accf[t][r]);
      }
    }
  }
}

// h = elu(acc/rowsum), re-encode to f16 padded to 224 for layer-2 gathers
__global__ void sg_finalize1(const float* __restrict__ acc, const float* __restrict__ rowsum,
                             _Float16* __restrict__ h16) {
  int i = blockIdx.x * 256 + threadIdx.x;
  if (i >= N_NODES * 224) return;
  int node = i / 224, k = i % 224;
  float v = 0.f;
  if (k < 200) {
    int h = k / 100;
    float rs = rowsum[node * 2 + h];
    rs = (rs == 0.f) ? 1e-12f : rs;
    v = elu1(acc[(size_t)node * 200 + k] / rs);
  }
  h16[i] = (_Float16)v;
}

// ---------------------------------------------------------------------------
// Layer-2 edge kernel: Npad=208 (13 tiles), K = 3 x 224 -> 21 WMMA k-steps.
// ---------------------------------------------------------------------------
__global__ __launch_bounds__(256) void sg_l2_edge(
    const _Float16* __restrict__ h16, const _Float16* __restrict__ ee16,
    const _Float16* __restrict__ B2, const float* __restrict__ a2p,
    const int* __restrict__ src, const int* __restrict__ dst,
    float* __restrict__ acc, float* __restrict__ rowsum) {
  const int lane = threadIdx.x & 31;
  const int strip = blockIdx.x * 8 + (threadIdx.x >> 5);
  if (strip >= N_STRIPS) return;
  const int e0 = strip * 16;
  const int hl = lane & 15, hi = lane >> 4;
  const int eA = e0 + hl;
  const int sA = src[eA], dA = dst[eA];
  const _Float16* arow[3] = { h16 + (size_t)sA * 224,
                              h16 + (size_t)dA * 224,
                              ee16 + (size_t)eA * 224 };
  v8f accf[13] = {};
#pragma unroll
  for (int seg = 0; seg < 3; ++seg) {
    const _Float16* ar = arow[seg];
#pragma unroll
    for (int kc = 0; kc < 7; ++kc) {
      v16h af = load_frag(ar, kc * 32, hi);
      const int kg = (seg * 7 + kc) * 32;
#pragma unroll
      for (int t = 0; t < 13; ++t) {
        v16h bf = load_frag(B2 + (size_t)(t * 16 + hl) * 672, kg, hi);
        accf[t] = __builtin_amdgcn_wmma_f32_16x16x32_f16(false, af, false, bf,
                                                         (short)0, accf[t], false, false);
      }
    }
  }
  int srcs[8];
#pragma unroll
  for (int r = 0; r < 8; ++r) srcs[r] = src[e0 + r + hi * 8];

  float p[8];
#pragma unroll
  for (int r = 0; r < 8; ++r) p[r] = 0.f;
#pragma unroll
  for (int t = 0; t < 13; ++t) {
    const float a2v = a2p[t * 16 + hl];
#pragma unroll
    for (int r = 0; r < 8; ++r) p[r] += accf[t][r] * a2v;
  }
#pragma unroll
  for (int m = 1; m < 16; m <<= 1)
#pragma unroll
    for (int r = 0; r < 8; ++r) p[r] += __shfl_xor(p[r], m, 32);
  float ev[8];
#pragma unroll
  for (int r = 0; r < 8; ++r) {
    float z = p[r];
    float lr = z > 0.f ? z : ALPHA_LR * z;
    ev[r] = expf(-lr);
  }
  if (hl == 0) {
#pragma unroll
    for (int r = 0; r < 8; ++r) atomAddF(&rowsum[srcs[r]], ev[r]);
  }
#pragma unroll
  for (int t = 0; t < 13; ++t) {
    const int cl = t * 16 + hl;
    if (cl < 200) {
#pragma unroll
      for (int r = 0; r < 8; ++r)
        atomAddF(&acc[(size_t)srcs[r] * 200 + cl], ev[r] * accf[t][r]);
    }
  }
}

__global__ void sg_final2(const float* __restrict__ acc, const float* __restrict__ rowsum,
                          float* __restrict__ out) {
  int i = blockIdx.x * 256 + threadIdx.x;
  if (i >= N_NODES * 200) return;
  int node = i / 200;
  float rs = rowsum[node];
  rs = (rs == 0.f) ? 1e-12f : rs;
  out[i] = elu1(acc[i] / rs);   // final elu(out_entity)
}

// ---------------------------------------------------------------------------
extern "C" void kernel_launch(void* const* d_in, const int* in_sizes, int n_in,
                              void* d_out, int out_size, void* d_ws, size_t ws_size,
                              hipStream_t stream) {
  const float* ent   = (const float*)d_in[0];
  const float* rel   = (const float*)d_in[1];
  const int*   el    = (const int*)d_in[2];
  const int*   etype = (const int*)d_in[3];
  const float* eemb  = (const float*)d_in[4];
  const int*   eln   = (const int*)d_in[5];
  const int*   etn   = (const int*)d_in[6];
  /* d_in[7] entity_embeddings_mapping unused by reference */
  const float* a_h   = (const float*)d_in[8];
  const float* a2_h  = (const float*)d_in[9];
  const float* W     = (const float*)d_in[10];
  const float* a_o   = (const float*)d_in[11];
  const float* a2_o  = (const float*)d_in[12];

  float* out_ent = (float*)d_out;                       // (30000, 200)
  float* out_rel = (float*)d_out + (size_t)N_NODES * 200; // (200, 200)

  char* ws = (char*)d_ws;
  _Float16* x16   = (_Float16*)(ws + X16_OFF);
  _Float16* h16   = (_Float16*)(ws + H16_OFF);
  _Float16* ee16  = (_Float16*)(ws + EE_OFF);   // ee1 then reused for ee2
  float*    acc   = (float*)(ws + ACC_OFF);
  float*    rs1   = (float*)(ws + RS1_OFF);
  float*    rs2   = (float*)(ws + RS2_OFF);
  _Float16* B1    = (_Float16*)(ws + B1_OFF);
  _Float16* B2    = (_Float16*)(ws + B2_OFF);
  _Float16* rel16 = (_Float16*)(ws + REL16_OFF);
  _Float16* WT16  = (_Float16*)(ws + WT16_OFF);
  float*    a2p1  = (float*)(ws + A2P1_OFF);
  float*    a2p2  = (float*)(ws + A2P2_OFF);
  int*      srcc  = (int*)(ws + SRC_OFF);
  int*      dstc  = (int*)(ws + DST_OFF);

  auto blk = [](size_t n) { return (unsigned)((n + 255) / 256); };

  // zero accumulators + both rowsums (contiguous region)
  hipMemsetAsync(ws + ACC_OFF, 0, ACC_SZ + RS1_SZ + RS2_SZ, stream);

  sg_build_edges<<<blk(N_EDGES), 256, 0, stream>>>(el, eln, srcc, dstc);
  sg_build_x16  <<<blk((size_t)N_NODES * 128), 256, 0, stream>>>(ent, x16);
  sg_build_ee1  <<<blk((size_t)N_EDGES * 128), 256, 0, stream>>>(eemb, rel, etn, ee16);
  sg_build_B1   <<<blk(224 * 384), 256, 0, stream>>>(a_h, B1);
  sg_build_a2p1 <<<1, 256, 0, stream>>>(a2_h, a2p1);
  sg_build_relWT<<<blk(208 * 128), 256, 0, stream>>>(rel, W, rel16, WT16);
  sg_relW_gemm  <<<blk(13 * 13 * 32), 256, 0, stream>>>(rel16, WT16, out_rel);
  sg_build_B2   <<<blk(208 * 672), 256, 0, stream>>>(a_o, B2);
  sg_build_a2p2 <<<1, 256, 0, stream>>>(a2_o, a2p2);

  // layer 1
  sg_l1_edge<<<blk((size_t)N_STRIPS * 32), 256, 0, stream>>>(
      x16, ee16, B1, a2p1, srcc, dstc, acc, rs1);
  sg_finalize1<<<blk((size_t)N_NODES * 224), 256, 0, stream>>>(acc, rs1, h16);

  // layer 2 (reuse acc buffer and ee buffer, stream-ordered)
  hipMemsetAsync(ws + ACC_OFF, 0, ACC_SZ, stream);
  sg_build_ee2<<<blk((size_t)N_EDGES * 224), 256, 0, stream>>>(out_rel, etype, etn, ee16);
  sg_l2_edge<<<blk((size_t)N_STRIPS * 32), 256, 0, stream>>>(
      h16, ee16, B2, a2p2, srcc, dstc, acc, rs2);
  sg_final2<<<blk((size_t)N_NODES * 200), 256, 0, stream>>>(acc, rs2, out_ent);
}
